// QuEn_6794638263052
// MI455X (gfx1250) — matrix operand order repaired and checked by reading, not compiled, not run on hardware
//
#include <hip/hip_runtime.h>
#include <hip/hip_bf16.h>

// ---------------------------------------------------------------------------
// Types for WMMA fragments (gfx1250, wave32)
// ---------------------------------------------------------------------------
typedef __attribute__((ext_vector_type(16))) _Float16 v16h;  // A/B fragment: 16x32 / 32x16 f16
typedef __attribute__((ext_vector_type(8)))  _Float16 h8;    // 16-byte half chunk
typedef __attribute__((ext_vector_type(8)))  float    v8f;   // C/D fragment: 16x16 f32

#define LDS_XT_STRIDE 72   // halves: 144 B rows -> 16B aligned, conflict-free B loads
#define LDS_T_STRIDE  68   // floats: 272 B rows -> 16B aligned, conflict-free A loads
#define NQ 64              // 2^6 state dimension
#define XT_BYTES (NQ * LDS_XT_STRIDE * 2)   // 9216 B per f16 X array
#define T_FLOATS (NQ * LDS_T_STRIDE)        // 4352 floats per T array

// ---------------------------------------------------------------------------
// Fragment loaders (layouts per CDNA5 ISA 7.12.2)
// ---------------------------------------------------------------------------

// A fragment (16x32 f16, MxK) from a row-major 64x64 half matrix.
// lane<16: M=row0+lane, halves 0..7 = K k0+0..7, halves 8..15 = K k0+16..23
// lane>=16: M=row0+lane-16, halves 0..7 = K k0+8..15, halves 8..15 = K k0+24..31
__device__ __forceinline__ v16h load_a_rowmajor(const _Float16* __restrict__ m,
                                                int row0, int k0, int l, int hi) {
  const _Float16* base = m + (row0 + l) * NQ;
  h8 lo  = *(const h8*)(base + k0 + hi * 8);
  h8 hi8 = *(const h8*)(base + k0 + 16 + hi * 8);
  return __builtin_shufflevector(lo, hi8, 0,1,2,3,4,5,6,7,8,9,10,11,12,13,14,15);
}

// B fragment (32x16 f16, KxN) where B[k][n] = m[n][k] (i.e. B = m^T), m row-major 64x64.
__device__ __forceinline__ v16h load_b_transposed(const _Float16* __restrict__ m,
                                                  int k0, int col0, int l, int hi) {
  const _Float16* p = m + (col0 + l) * NQ + k0 + hi * 8;
  h8 lo  = *(const h8*)p;
  h8 hi8 = *(const h8*)(p + 16);
  return __builtin_shufflevector(lo, hi8, 0,1,2,3,4,5,6,7,8,9,10,11,12,13,14,15);
}

// B fragment from LDS array holding X^T (padded stride), B[k][n] = X[k][n] = Xt[n][k].
__device__ __forceinline__ v16h load_b_lds_xt(const _Float16* s,
                                              int k0, int col0, int l, int hi) {
  const _Float16* p = s + (col0 + l) * LDS_XT_STRIDE + k0 + hi * 8;
  h8 lo  = *(const h8*)p;
  h8 hi8 = *(const h8*)(p + 16);
  return __builtin_shufflevector(lo, hi8, 0,1,2,3,4,5,6,7,8,9,10,11,12,13,14,15);
}

// Split A fragment from a row-major fp32 LDS matrix: ah = f16(v), al = f16(v - ah).
__device__ __forceinline__ void load_a_lds_split(const float* s, int row0, int k0,
                                                 int l, int hi, v16h& ah, v16h& al) {
  const float* p = s + (row0 + l) * LDS_T_STRIDE;
#pragma unroll
  for (int j = 0; j < 8; ++j) {
    float v0 = p[k0 + hi * 8 + j];
    _Float16 h0 = (_Float16)v0;
    ah[j] = h0;  al[j] = (_Float16)(v0 - (float)h0);
    float v1 = p[k0 + 16 + hi * 8 + j];
    _Float16 h1 = (_Float16)v1;
    ah[8 + j] = h1;  al[8 + j] = (_Float16)(v1 - (float)h1);
  }
}

__device__ __forceinline__ v8f wmma_f16(v16h a, v16h b, v8f c) {
  return __builtin_amdgcn_wmma_f32_16x16x32_f16(false, a, false, b,
                                                (short)0, c, false, false);
}

// ---------------------------------------------------------------------------
// Kernel 1: build U from the 18 rotation parameters, store two-term f16 split.
// One block of 64 threads; thread c owns column c (gates are row ops -> columns
// independent, no barriers).
// ---------------------------------------------------------------------------
__global__ __launch_bounds__(64) void QuEn_build_u(
    const float* __restrict__ weight, _Float16* __restrict__ W)
{
  __shared__ float Ur[NQ][NQ];
  __shared__ float Ui[NQ][NQ];
  const int c = threadIdx.x;

  for (int r = 0; r < NQ; ++r) { Ur[r][c] = (r == c) ? 1.0f : 0.0f; Ui[r][c] = 0.0f; }

  const float W_MUL = 0.6324555320336759f;  // sqrt(2)/sqrt(5)
  float w[18];
#pragma unroll
  for (int i = 0; i < 18; ++i) w[i] = weight[i] * W_MUL;

  auto rx = [&](int q, float t) {
    float cs = cosf(0.5f * t), sn = sinf(0.5f * t);
    int st = 1 << (5 - q);
    for (int r0 = 0; r0 < NQ; ++r0) if (!(r0 & st)) {
      int r1 = r0 | st;
      float ar = Ur[r0][c], ai = Ui[r0][c], br = Ur[r1][c], bi = Ui[r1][c];
      Ur[r0][c] = cs * ar + sn * bi;  Ui[r0][c] = cs * ai - sn * br;
      Ur[r1][c] = sn * ai + cs * br;  Ui[r1][c] = cs * bi - sn * ar;
    }
  };
  auto ry = [&](int q, float t) {
    float cs = cosf(0.5f * t), sn = sinf(0.5f * t);
    int st = 1 << (5 - q);
    for (int r0 = 0; r0 < NQ; ++r0) if (!(r0 & st)) {
      int r1 = r0 | st;
      float ar = Ur[r0][c], ai = Ui[r0][c], br = Ur[r1][c], bi = Ui[r1][c];
      Ur[r0][c] = cs * ar - sn * br;  Ui[r0][c] = cs * ai - sn * bi;
      Ur[r1][c] = sn * ar + cs * br;  Ui[r1][c] = sn * ai + cs * bi;
    }
  };
  auto rz = [&](int q, float t) {
    float cs = cosf(0.5f * t), sn = sinf(0.5f * t);  // e^{-it/2} = (cs, -sn)
    int st = 1 << (5 - q);
    for (int r0 = 0; r0 < NQ; ++r0) if (!(r0 & st)) {
      int r1 = r0 | st;
      float ar = Ur[r0][c], ai = Ui[r0][c], br = Ur[r1][c], bi = Ui[r1][c];
      Ur[r0][c] = cs * ar + sn * ai;  Ui[r0][c] = cs * ai - sn * ar;
      Ur[r1][c] = cs * br - sn * bi;  Ui[r1][c] = cs * bi + sn * br;
    }
  };
  auto cnot = [&](int q) {  // control q-1, target q
    int cb = 1 << (6 - q), tb = 1 << (5 - q);
    for (int r = 0; r < NQ; ++r) if ((r & cb) && !(r & tb)) {
      int r2 = r | tb;
      float tr = Ur[r][c]; Ur[r][c] = Ur[r2][c]; Ur[r2][c] = tr;
      float ti = Ui[r][c]; Ui[r][c] = Ui[r2][c]; Ui[r2][c] = ti;
    }
  };

  for (int q = 0; q < 6; ++q) { rx(q, w[q]); ry(q, w[q + 6]); rz(q, w[q + 12]); }
  for (int q = 1; q < 6; ++q) cnot(q);
  for (int q = 0; q < 6; ++q) { rx(q, -w[q]); ry(q, -w[q + 6]); rz(q, -w[q + 12]); }

  // Two-term split: Ph + Pl ~= Re(U) to ~2^-22, same for Q; Qn* = -Q* (exact).
  _Float16* Ph  = W;
  _Float16* Pl  = W + 1 * NQ * NQ;
  _Float16* Qh  = W + 2 * NQ * NQ;
  _Float16* Ql  = W + 3 * NQ * NQ;
  _Float16* Qnh = W + 4 * NQ * NQ;
  _Float16* Qnl = W + 5 * NQ * NQ;
  for (int r = 0; r < NQ; ++r) {
    float pr = Ur[r][c], qi = Ui[r][c];
    _Float16 ph = (_Float16)pr;  _Float16 pl = (_Float16)(pr - (float)ph);
    _Float16 qh = (_Float16)qi;  _Float16 ql = (_Float16)(qi - (float)qh);
    Ph[r * NQ + c]  = ph;   Pl[r * NQ + c]  = pl;
    Qh[r * NQ + c]  = qh;   Ql[r * NQ + c]  = ql;
    Qnh[r * NQ + c] = -qh;  Qnl[r * NQ + c] = -ql;
  }
}

// ---------------------------------------------------------------------------
// Kernel 2: out = U X U^dag per batch element, two-term f16 split arithmetic.
//   T  = U X :      Tr = P.R + Qn.S        Ti = P.S + Q.R
//   D  = T U^dag:   Dr = Tr.P^T + Ti.Q^T   Di = Ti.P^T + Tr.Qn^T
// Each product M.N expands to Mh.Nh + Mh.Nl + Ml.Nh (Ml.Nl ~ 2^-22 dropped),
// giving fp32-grade accuracy at full f16 WMMA rate (still under the HBM
// roofline: ~51 GFLOP total vs 268 MB of traffic).
// One block (8 wave32) per batch element; wave owns row-strip ti and two
// adjacent column tiles (A-fragments reused across both). X-split arrays and
// fp32 T arrays alias the same LDS buffer (barrier-separated phases).
// ---------------------------------------------------------------------------
__global__ __launch_bounds__(256, 1) void QuEn_conj_density(
    const float* __restrict__ x_re, const float* __restrict__ x_im,
    const _Float16* __restrict__ W, float* __restrict__ out)
{
  __shared__ __align__(16) char smem[4 * XT_BYTES];  // 36864 B, dual-purpose

  _Float16* Rh = (_Float16*)smem;                    // X_re hi, transposed
  _Float16* Rl = (_Float16*)(smem + 1 * XT_BYTES);   // X_re lo
  _Float16* Sh = (_Float16*)(smem + 2 * XT_BYTES);   // X_im hi
  _Float16* Sl = (_Float16*)(smem + 3 * XT_BYTES);   // X_im lo
  float* TrS = (float*)smem;                         // Re(U X), fp32 (aliases X)
  float* TiS = TrS + T_FLOATS;                       // Im(U X), fp32

  const _Float16* Ph  = W;
  const _Float16* Pl  = W + 1 * NQ * NQ;
  const _Float16* Qh  = W + 2 * NQ * NQ;
  const _Float16* Ql  = W + 3 * NQ * NQ;
  const _Float16* Qnh = W + 4 * NQ * NQ;
  const _Float16* Qnl = W + 5 * NQ * NQ;

  const int b    = blockIdx.x;
  const int tid  = threadIdx.x;
  const int wave = tid >> 5;
  const int lane = tid & 31;
  const int l    = lane & 15;
  const int hi   = lane >> 4;
  const int ti   = wave >> 1;        // row-strip of this wave
  const int tc0  = (wave & 1) * 2;   // first of two column tiles

  const float* xr = x_re + (size_t)b * (NQ * NQ);
  const float* xi = x_im + (size_t)b * (NQ * NQ);

  // ---- Load X (fp32), two-term split, store transposed f16 into LDS ----
  for (int ch = tid; ch < 1024; ch += 256) {   // 1024 float4 chunks per matrix
    int r  = ch >> 4;                          // source row
    int c0 = (ch & 15) << 2;                   // source col base
    float4 vr = ((const float4*)xr)[ch];
    float4 vi = ((const float4*)xi)[ch];
    float re[4] = {vr.x, vr.y, vr.z, vr.w};
    float im[4] = {vi.x, vi.y, vi.z, vi.w};
#pragma unroll
    for (int j = 0; j < 4; ++j) {
      int o = (c0 + j) * LDS_XT_STRIDE + r;
      _Float16 rh = (_Float16)re[j];
      _Float16 ih = (_Float16)im[j];
      Rh[o] = rh;  Rl[o] = (_Float16)(re[j] - (float)rh);
      Sh[o] = ih;  Sl[o] = (_Float16)(im[j] - (float)ih);
    }
  }
  __syncthreads();

  // ---- Stage 1: T = U X (accumulators stay in registers across the barrier) ----
  v8f acc[4] = {v8f{}, v8f{}, v8f{}, v8f{}};   // [tile][re/im] = acc[2*t + c]
#pragma unroll
  for (int ks = 0; ks < 2; ++ks) {
    const int k0 = ks * 32;
    v16h aPh  = load_a_rowmajor(Ph,  ti * 16, k0, l, hi);
    v16h aPl  = load_a_rowmajor(Pl,  ti * 16, k0, l, hi);
    v16h aQh  = load_a_rowmajor(Qh,  ti * 16, k0, l, hi);
    v16h aQl  = load_a_rowmajor(Ql,  ti * 16, k0, l, hi);
    v16h aQnh = load_a_rowmajor(Qnh, ti * 16, k0, l, hi);
    v16h aQnl = load_a_rowmajor(Qnl, ti * 16, k0, l, hi);
#pragma unroll
    for (int t = 0; t < 2; ++t) {
      const int col = (tc0 + t) * 16;
      v16h bRh = load_b_lds_xt(Rh, k0, col, l, hi);
      v16h bRl = load_b_lds_xt(Rl, k0, col, l, hi);
      v16h bSh = load_b_lds_xt(Sh, k0, col, l, hi);
      v16h bSl = load_b_lds_xt(Sl, k0, col, l, hi);
      v8f tr = acc[2 * t], tii = acc[2 * t + 1];
      tr  = wmma_f16(aPh,  bRh, tr);   // P.R  (hi.hi + hi.lo + lo.hi)
      tr  = wmma_f16(aPh,  bRl, tr);
      tr  = wmma_f16(aPl,  bRh, tr);
      tr  = wmma_f16(aQnh, bSh, tr);   // -Q.S
      tr  = wmma_f16(aQnh, bSl, tr);
      tr  = wmma_f16(aQnl, bSh, tr);
      tii = wmma_f16(aPh,  bSh, tii);  // P.S
      tii = wmma_f16(aPh,  bSl, tii);
      tii = wmma_f16(aPl,  bSh, tii);
      tii = wmma_f16(aQh,  bRh, tii);  // Q.R
      tii = wmma_f16(aQh,  bRl, tii);
      tii = wmma_f16(aQl,  bRh, tii);
      acc[2 * t] = tr;  acc[2 * t + 1] = tii;
    }
  }
  __syncthreads();   // all X reads done -> safe to overwrite aliased LDS with T

#pragma unroll
  for (int t = 0; t < 2; ++t) {
#pragma unroll
    for (int v = 0; v < 8; ++v) {
      const int row = ti * 16 + v + 8 * hi;
      const int col = (tc0 + t) * 16 + l;
      TrS[row * LDS_T_STRIDE + col] = acc[2 * t][v];
      TiS[row * LDS_T_STRIDE + col] = acc[2 * t + 1][v];
    }
  }
  __syncthreads();

  // ---- Stage 2: D = T U^dag, write straight to global ----
  float* outr = out + (size_t)b * (NQ * NQ);
  float* outi = outr + (size_t)4096 * (NQ * NQ);  // imag plane offset = B*64*64
  v8f oacc[4] = {v8f{}, v8f{}, v8f{}, v8f{}};
#pragma unroll
  for (int ks = 0; ks < 2; ++ks) {
    const int k0 = ks * 32;
    v16h aTrh, aTrl, aTih, aTil;
    load_a_lds_split(TrS, ti * 16, k0, l, hi, aTrh, aTrl);
    load_a_lds_split(TiS, ti * 16, k0, l, hi, aTih, aTil);
#pragma unroll
    for (int t = 0; t < 2; ++t) {
      const int col = (tc0 + t) * 16;
      v16h bPh  = load_b_transposed(Ph,  k0, col, l, hi);
      v16h bPl  = load_b_transposed(Pl,  k0, col, l, hi);
      v16h bQh  = load_b_transposed(Qh,  k0, col, l, hi);
      v16h bQl  = load_b_transposed(Ql,  k0, col, l, hi);
      v16h bQnh = load_b_transposed(Qnh, k0, col, l, hi);
      v16h bQnl = load_b_transposed(Qnl, k0, col, l, hi);
      v8f dr = oacc[2 * t], di = oacc[2 * t + 1];
      dr = wmma_f16(aTrh, bPh,  dr);   // Tr.P^T
      dr = wmma_f16(aTrh, bPl,  dr);
      dr = wmma_f16(aTrl, bPh,  dr);
      dr = wmma_f16(aTih, bQh,  dr);   // Ti.Q^T
      dr = wmma_f16(aTih, bQl,  dr);
      dr = wmma_f16(aTil, bQh,  dr);
      di = wmma_f16(aTih, bPh,  di);   // Ti.P^T
      di = wmma_f16(aTih, bPl,  di);
      di = wmma_f16(aTil, bPh,  di);
      di = wmma_f16(aTrh, bQnh, di);   // -Tr.Q^T
      di = wmma_f16(aTrh, bQnl, di);
      di = wmma_f16(aTrl, bQnh, di);
      oacc[2 * t] = dr;  oacc[2 * t + 1] = di;
    }
  }
#pragma unroll
  for (int t = 0; t < 2; ++t) {
#pragma unroll
    for (int v = 0; v < 8; ++v) {
      const int row = ti * 16 + v + 8 * hi;
      const int col = (tc0 + t) * 16 + l;
      outr[row * NQ + col] = oacc[2 * t][v];
      outi[row * NQ + col] = oacc[2 * t + 1][v];
    }
  }
}

// ---------------------------------------------------------------------------
// Launcher
// ---------------------------------------------------------------------------
extern "C" void kernel_launch(void* const* d_in, const int* in_sizes, int n_in,
                              void* d_out, int out_size, void* d_ws, size_t ws_size,
                              hipStream_t stream) {
  const float* x_re   = (const float*)d_in[0];
  const float* x_im   = (const float*)d_in[1];
  const float* weight = (const float*)d_in[2];
  float* out = (float*)d_out;

  _Float16* W = (_Float16*)d_ws;   // 6 * 64*64 halves (Ph,Pl,Qh,Ql,Qnh,Qnl)

  QuEn_build_u<<<1, 64, 0, stream>>>(weight, W);
  QuEn_conj_density<<<4096, 256, 0, stream>>>(x_re, x_im, W, out);
}
